// StateBank_2095944040927
// MI455X (gfx1250) — matrix operand order repaired and checked by reading, not compiled, not run on hardware
//
#include <hip/hip_runtime.h>
#include <hip/hip_bf16.h>

typedef unsigned short u16;
typedef __attribute__((ext_vector_type(16))) __bf16 v16bf;
typedef __attribute__((ext_vector_type(8)))  float  v8f;
typedef __attribute__((ext_vector_type(4)))  unsigned int u32x4;
typedef __attribute__((ext_vector_type(8)))  int i32x8;
typedef __attribute__((ext_vector_type(4)))  int i32x4;

union ABFrag { v16bf v; uint4 u4[2]; };

#define DDIM 1024
#define ROWS 16
#define TOPK 32

#if __has_builtin(__builtin_amdgcn_tensor_load_to_lds)
#define HAVE_TDM 1
#else
#define HAVE_TDM 0
#endif

// ---------------- fp32 -> bf16 (round to nearest even) ----------------
__global__ void f32_to_bf16_kernel(const float* __restrict__ src,
                                   u16* __restrict__ dst, int n) {
    int i = blockIdx.x * blockDim.x + threadIdx.x;
    int stride = gridDim.x * blockDim.x;
    for (; i < n; i += stride) {
        union { float f; unsigned u; } x;
        x.f = src[i];
        unsigned u = x.u;
        unsigned r = (u + 0x7FFFu + ((u >> 16) & 1u)) >> 16;
        dst[i] = (u16)r;
    }
}

__global__ void zero_f32_kernel(float* __restrict__ p, int n) {
    int i = blockIdx.x * blockDim.x + threadIdx.x;
    if (i < n) p[i] = 0.0f;
}

#if HAVE_TDM
// Issue a TDM load of 16x1024 bf16 (row-major) starting at key row sBase
// into LDS at ldsDst. Descriptor per CDNA5 ISA.
__device__ __forceinline__ void tdm_load_chunk(const u16* kb, int S, int sBase,
                                               u16* ldsDst) {
    unsigned long long ga = (unsigned long long)(uintptr_t)(kb + (size_t)sBase * DDIM);
    unsigned ldsOff = (unsigned)(uintptr_t)(void*)ldsDst;  // low 32 bits = LDS byte offset

    u32x4 g0;
    g0[0] = 1u;                                   // count=1, user descriptor
    g0[1] = ldsOff;                               // lds_addr (bytes)
    g0[2] = (unsigned)(ga & 0xFFFFFFFFu);         // global_addr[31:0]
    g0[3] = ((unsigned)(ga >> 32) & 0x01FFFFFFu)  // global_addr[56:32]
            | (2u << 30);                         // type = 2 ("image")

    i32x8 g1;
    g1[0] = (int)(1u << 16);                          // data_size=1 -> 2 bytes/elem
    g1[1] = (int)((unsigned)(DDIM & 0xFFFF) << 16);   // tensor_dim0[15:0] = 1024
    g1[2] = (int)(((unsigned)(S - sBase) & 0xFFFFu) << 16); // tensor_dim1[15:0]
    g1[3] = (int)((unsigned)(DDIM & 0xFFFF) << 16);   // tile_dim0 = 1024
    g1[4] = ROWS;                                     // tile_dim1 = 16
    g1[5] = DDIM;                                     // tensor_dim0_stride = 1024
    g1[6] = 0;
    g1[7] = 0;

    i32x4 z4 = {0, 0, 0, 0};
    i32x8 z8 = {0, 0, 0, 0, 0, 0, 0, 0};
    __builtin_amdgcn_tensor_load_to_lds(g0, g1, z4, z4, z8, 0);
}
#endif

// ---------------- hierarchical top-k attention, one level per block.y ----
__global__ __launch_bounds__(32) void hier_topk_attn(
    const u16* __restrict__ qb,
    const u16* __restrict__ k0b, const u16* __restrict__ k1b, const u16* __restrict__ k2b,
    const float* __restrict__ v0, const float* __restrict__ v1, const float* __restrict__ v2,
    const float* __restrict__ s0, const float* __restrict__ s1, const float* __restrict__ s2,
    float* __restrict__ out)
{
    __shared__ u16   qtile[ROWS * DDIM];     // 32 KB: Q tile in bf16
    __shared__ u16   kbuf[2][ROWS * DDIM];   // 64 KB: double-buffered K chunks
    __shared__ float scoreT[ROWS][ROWS];
    __shared__ float topS[ROWS][TOPK];
    __shared__ int   topI[ROWS][TOPK];
    __shared__ float wts[ROWS][TOPK];

    const int lane = threadIdx.x;            // 0..31 (wave32)
    const int tile = blockIdx.x;             // 0..511
    const int lvl  = blockIdx.y;             // 0..2

    const u16*   kb  = (lvl == 0) ? k0b : (lvl == 1) ? k1b : k2b;
    const float* vv  = (lvl == 0) ? v0  : (lvl == 1) ? v1  : v2;
    const float* sal = (lvl == 0) ? s0  : (lvl == 1) ? s1  : s2;
    const int    S   = (lvl == 0) ? 8192 : (lvl == 1) ? 4096 : 2048;

    const int qrow0 = tile * ROWS;

#if HAVE_TDM
    // kick off the first K chunk on the Tensor Data Mover ASAP
    tdm_load_chunk(kb, S, 0, &kbuf[0][0]);
#endif

    // ---- stage Q tile (contiguous 32 KB) into LDS ----
    {
        const uint4* src = (const uint4*)(qb + (size_t)qrow0 * DDIM);
        uint4*       dst = (uint4*)qtile;
        for (int i = lane; i < (ROWS * DDIM) / 8; i += 32)
            dst[i] = src[i];
    }
    // ---- init top-k lists ----
    for (int i = lane; i < ROWS * TOPK; i += 32) {
        ((float*)topS)[i] = -__builtin_inff();
        ((int*)topI)[i]   = 0;
    }
    __syncthreads();

    const int colN = lane & 15;    // B column == key row within chunk; C column
    const int half = lane >> 4;    // lane half selects K sub-range of fragments
    const int aK   = half * 8;     // A: K base of first 8 elements (then +16)
    const int bK   = half * 16;    // unused; B loads use aK split like A

    float rowMin = -__builtin_inff();
    int   rowArg = 0;
    const float scale = 0.03125f;  // 1/sqrt(1024)

    const int nchunk = S / 16;
    for (int ci = 0; ci < nchunk; ++ci) {
        const int sBase = ci * 16;

#if HAVE_TDM
        // double-buffer: TDM streams chunk ci+1 while we compute chunk ci
        if (ci + 1 < nchunk) {
            tdm_load_chunk(kb, S, sBase + 16, &kbuf[(ci + 1) & 1][0]);
            __builtin_amdgcn_s_wait_tensorcnt(1);   // oldest (chunk ci) complete
        } else {
            __builtin_amdgcn_s_wait_tensorcnt(0);
        }
        __syncthreads();                            // order LDS reads after TDM
        const u16* krow = &kbuf[ci & 1][colN * DDIM];
#else
        const u16* krow = kb + (size_t)(sBase + colN) * DDIM;
        if (sBase + 16 < S)
            __builtin_prefetch(kb + (size_t)(sBase + 16 + colN) * DDIM + half * 512, 0, 1);
#endif

        // ---- 16x16 score tile: 32 accumulating bf16 WMMAs over D=1024 ----
        v8f acc = {};
        #pragma unroll 4
        for (int dc = 0; dc < DDIM / 32; ++dc) {
            ABFrag a, b;
            const int abase = colN * DDIM + dc * 32 + aK;
            a.u4[0] = *(const uint4*)&qtile[abase];       // K = aK .. aK+7
            a.u4[1] = *(const uint4*)&qtile[abase + 16];  // K = aK+16 .. aK+23
            const int bbase = dc * 32 + aK;
            b.u4[0] = *(const uint4*)&krow[bbase];        // K = aK .. aK+7
            b.u4[1] = *(const uint4*)&krow[bbase + 16];   // K = aK+16 .. aK+23
            acc = __builtin_amdgcn_wmma_f32_16x16x32_bf16(
                false, a.v, false, b.v, (short)0, acc, false, false);
        }

        // ---- scale + salience, scatter score tile to LDS ----
        {
            const float sv = sal[sBase + colN];
            #pragma unroll
            for (int m = 0; m < 8; ++m)
                scoreT[m + half * 8][colN] = acc[m] * scale + sv;
        }
        __syncthreads();

        // ---- per-row top-32 maintenance: lane r owns row r ----
        if (lane < ROWS) {
            const int r = lane;
            #pragma unroll
            for (int j = 0; j < 16; ++j) {
                float c = scoreT[r][j];
                if (c > rowMin) {
                    topS[r][rowArg] = c;
                    topI[r][rowArg] = sBase + j;
                    float m = topS[r][0]; int am = 0;
                    #pragma unroll 8
                    for (int k = 1; k < TOPK; ++k) {
                        float t = topS[r][k];
                        if (t < m) { m = t; am = k; }
                    }
                    rowMin = m; rowArg = am;
                }
            }
        }
        __syncthreads();
    }

    // ---- softmax over each row's 32 survivors ----
    if (lane < ROWS) {
        const int r = lane;
        float mx = -__builtin_inff();
        for (int k = 0; k < TOPK; ++k) mx = fmaxf(mx, topS[r][k]);
        float sum = 0.0f;
        for (int k = 0; k < TOPK; ++k) {
            float e = __expf(topS[r][k] - mx);
            wts[r][k] = e;
            sum += e;
        }
        float inv = 1.0f / sum;
        for (int k = 0; k < TOPK; ++k) wts[r][k] *= inv;
    }
    __syncthreads();

    // ---- gather V (fp32, L2-resident) + weighted sum; atomic add across levels
    for (int r = 0; r < ROWS; ++r) {
        float4 accv[8];
        #pragma unroll
        for (int j = 0; j < 8; ++j) accv[j] = make_float4(0.f, 0.f, 0.f, 0.f);
        for (int k = 0; k < TOPK; ++k) {
            const float w = wts[r][k];
            const float4* vr = (const float4*)(vv + (size_t)topI[r][k] * DDIM);
            #pragma unroll
            for (int j = 0; j < 8; ++j) {
                float4 x = vr[lane + 32 * j];
                accv[j].x += w * x.x; accv[j].y += w * x.y;
                accv[j].z += w * x.z; accv[j].w += w * x.w;
            }
        }
        float* orow = out + (size_t)(qrow0 + r) * DDIM;
        #pragma unroll
        for (int j = 0; j < 8; ++j) {
            const int d = (lane + 32 * j) * 4;
            atomicAdd(&orow[d + 0], accv[j].x);
            atomicAdd(&orow[d + 1], accv[j].y);
            atomicAdd(&orow[d + 2], accv[j].z);
            atomicAdd(&orow[d + 3], accv[j].w);
        }
    }
}

extern "C" void kernel_launch(void* const* d_in, const int* in_sizes, int n_in,
                              void* d_out, int out_size, void* d_ws, size_t ws_size,
                              hipStream_t stream) {
    const float* q  = (const float*)d_in[0];
    const float* k0 = (const float*)d_in[1];
    const float* v0 = (const float*)d_in[2];
    const float* s0 = (const float*)d_in[3];
    const float* k1 = (const float*)d_in[4];
    const float* v1 = (const float*)d_in[5];
    const float* s1 = (const float*)d_in[6];
    const float* k2 = (const float*)d_in[7];
    const float* v2 = (const float*)d_in[8];
    const float* s2 = (const float*)d_in[9];
    float* out = (float*)d_out;

    // workspace layout (bf16 copies of Q and the three key banks)
    const int NQ  = 8192 * 1024;
    const int NK0 = 8192 * 1024;
    const int NK1 = 4096 * 1024;
    const int NK2 = 2048 * 1024;
    u16* qb  = (u16*)d_ws;
    u16* k0b = qb  + NQ;
    u16* k1b = k0b + NK0;
    u16* k2b = k1b + NK1;

    f32_to_bf16_kernel<<<2048, 256, 0, stream>>>(q,  qb,  NQ);
    f32_to_bf16_kernel<<<2048, 256, 0, stream>>>(k0, k0b, NK0);
    f32_to_bf16_kernel<<<1024, 256, 0, stream>>>(k1, k1b, NK1);
    f32_to_bf16_kernel<<< 512, 256, 0, stream>>>(k2, k2b, NK2);
    zero_f32_kernel<<<(NQ + 255) / 256, 256, 0, stream>>>(out, NQ);

    dim3 grid(512, 3);   // 512 query tiles of 16 rows x 3 levels
    hier_topk_attn<<<grid, 32, 0, stream>>>(qb, k0b, k1b, k2b,
                                            v0, v1, v2, s0, s1, s2, out);
}
